// QMixtralBlockSparseTop2MLP_30820685316560
// MI455X (gfx1250) — matrix operand order repaired and checked by reading, not compiled, not run on hardware
//
#include <hip/hip_runtime.h>

// Problem sizes (fixed by the reference)
#define Q_LEN  2048
#define HIDDEN 4096
#define FFN    14336

typedef __attribute__((ext_vector_type(16))) _Float16 v16h;
typedef __attribute__((ext_vector_type(8)))  _Float16 v8h;
typedef __attribute__((ext_vector_type(4)))  _Float16 v4h;
typedef __attribute__((ext_vector_type(8)))  float    v8f;

// Block tile: 128M x 128N x 32K, 8 wave32 per block, each wave = 16M x 128N.
#define BM 128
#define BN 128
#define BK 32
#define LDK 40   // padded K stride (halves): 80B rows avoid 4-way LDS bank conflicts

__device__ __forceinline__ v16h join16(v8h lo, v8h hi) {
  return __builtin_shufflevector(lo, hi, 0,1,2,3,4,5,6,7,8,9,10,11,12,13,14,15);
}

// A fragment (16x32 f16), CDNA5 wave32 layout:
// lane L: M = L%16; lanes 0-15 hold K in {0..7, 16..23}, lanes 16-31 hold {8..15, 24..31}.
__device__ __forceinline__ v16h load_a_frag(const _Float16* rowBase, int lane) {
  const _Float16* p = rowBase + (lane & 15) * LDK + ((lane >> 4) * 8);
  return join16(*(const v8h*)p, *(const v8h*)(p + 16));
}

// B fragment (32x16 f16): lane L: N = L%16; lanes 0-15 hold K in {0..7, 16..23}, lanes 16-31 hold {8..15, 24..31}.
__device__ __forceinline__ v16h load_b_frag(const _Float16* rowBase, int lane) {
  const _Float16* p = rowBase + (lane & 15) * LDK + ((lane >> 4) * 8);
  return join16(*(const v8h*)p, *(const v8h*)(p + 16));
}

// int8 code (stored as int32) -> exact f16 (q-128), 4 at a time, one b64 LDS store.
__device__ __forceinline__ void cvt_store_q(_Float16* dst, int4 q) {
  v4h t; t[0] = (_Float16)(q.x - 128); t[1] = (_Float16)(q.y - 128);
         t[2] = (_Float16)(q.z - 128); t[3] = (_Float16)(q.w - 128);
  *(v4h*)dst = t;
}

// 16B global -> LDS async copy (ASYNCcnt-tracked, no VGPR data path).
__device__ __forceinline__ void async_copy_b128(const void* gptr, void* lptr) {
  const unsigned lds = (unsigned)(size_t)lptr;             // low 32b = LDS offset
  const unsigned long long g = (unsigned long long)(size_t)gptr;
  asm volatile("global_load_async_to_lds_b128 %0, %1, off"
               :: "v"(lds), "v"(g) : "memory");
}

#define WMMA_F16(A, B, C) \
  __builtin_amdgcn_wmma_f32_16x16x32_f16(false, (A), false, (B), (short)0, (C), false, false)

// ---------------------------------------------------------------------------
// Kernel 0: one-time x fp32 -> f16 conversion (K-invariant work hoisted out
// of the 112-fold reuse inside gateup). 16.8 MB result lives in d_ws.
// ---------------------------------------------------------------------------
__global__ void __launch_bounds__(256)
cvt_x_kernel(const float* __restrict__ x, _Float16* __restrict__ xh)
{
  const size_t i = ((size_t)blockIdx.x * 256 + threadIdx.x) * 8;
  const float4 a = *(const float4*)(x + i);
  const float4 b = *(const float4*)(x + i + 4);
  v8h t;
  t[0] = (_Float16)a.x; t[1] = (_Float16)a.y; t[2] = (_Float16)a.z; t[3] = (_Float16)a.w;
  t[4] = (_Float16)b.x; t[5] = (_Float16)b.y; t[6] = (_Float16)b.z; t[7] = (_Float16)b.w;
  *(v8h*)(xh + i) = t;
}

// ---------------------------------------------------------------------------
// Kernel 1: h[m,n] = silu(s1[n]*sum_k x[m,k]*(q1[n,k]-128)) *
//                    (s3[n]*sum_k x[m,k]*(q3[n,k]-128)),  stored f16.
// Double-buffered LDS, ONE barrier per K-step. A tile streams via async
// global->LDS; weight tiles are reg-staged + dequantized on the store path
// (VALU co-executes with TRANS-tracked f16 WMMAs).
// ---------------------------------------------------------------------------
__global__ void __launch_bounds__(256)
gateup_kernel(const _Float16* __restrict__ xh,
              const int* __restrict__ w1q, const float* __restrict__ s1,
              const int* __restrict__ w3q, const float* __restrict__ s3,
              _Float16* __restrict__ h)
{
  __shared__ _Float16 lA [2][BM][LDK];
  __shared__ _Float16 lB1[2][BN][LDK];
  __shared__ _Float16 lB3[2][BN][LDK];

  const int tid  = threadIdx.x;
  const int lane = tid & 31;
  const int wave = tid >> 5;              // 0..7 -> 16-row M strip
  const int m0   = blockIdx.x * BM;
  const int n0   = blockIdx.y * BN;

  // Weight staging: 32 rows per pass, 8 threads/row, 16B per thread.
  const int srow = tid >> 3;              // 0..31
  const int scol = (tid & 7) * 4;         // 0,4,...,28
  // A-tile async staging: 64 rows per pass, 4 threads/row, 16B per thread.
  const int arow = tid >> 2;              // 0..63
  const int acol = (tid & 3) * 8;         // halves

  const _Float16* pa = xh  + (size_t)(m0 + arow) * HIDDEN + acol;
  const int*      p1 = w1q + (size_t)(n0 + srow) * HIDDEN + scol;
  const int*      p3 = w3q + (size_t)(n0 + srow) * HIDDEN + scol;
  const size_t astep = (size_t)64 * HIDDEN;
  const size_t wstep = (size_t)32 * HIDDEN;

  v8f accG[8] = {};
  v8f accU[8] = {};

  auto issue_async = [&](int buf, int k0) {
    #pragma unroll
    for (int p = 0; p < 2; ++p)
      async_copy_b128(pa + p * astep + k0, &lA[buf][arow + p * 64][acol]);
  };

  // Pipeline prologue: tile 0.
  int4 r1[4], r3[4];
  issue_async(0, 0);
  #pragma unroll
  for (int p = 0; p < 4; ++p) {
    r1[p] = *(const int4*)(p1 + p * wstep);
    r3[p] = *(const int4*)(p3 + p * wstep);
  }

  const int NK = HIDDEN / BK;
  for (int kt = 0; kt < NK; ++kt) {
    const int buf = kt & 1;
    // Dequant staged weight registers into LDS (exact: codes-128 fit f16).
    #pragma unroll
    for (int p = 0; p < 4; ++p) {
      cvt_store_q(&lB1[buf][srow + p * 32][scol], r1[p]);
      cvt_store_q(&lB3[buf][srow + p * 32][scol], r3[p]);
    }
    // Issue next tile's loads now; they land during the WMMAs below.
    if (kt + 1 < NK) {
      const int k0 = (kt + 1) * BK;
      issue_async(buf ^ 1, k0);
      #pragma unroll
      for (int p = 0; p < 4; ++p) {
        r1[p] = *(const int4*)(p1 + p * wstep + k0);
        r3[p] = *(const int4*)(p3 + p * wstep + k0);
        __builtin_prefetch(p1 + p * wstep + k0 + BK, 0, 1);
        __builtin_prefetch(p3 + p * wstep + k0 + BK, 0, 1);
      }
      asm volatile("s_wait_asynccnt 0x2" ::: "memory"); // tile kt's A landed
    } else {
      asm volatile("s_wait_asynccnt 0x0" ::: "memory");
    }
    __syncthreads();   // single barrier: publishes buf, retires buf^1 readers

    const v16h af = load_a_frag(&lA[buf][wave * 16][0], lane);
    v16h b1 = load_b_frag(&lB1[buf][0][0], lane);
    v16h b3 = load_b_frag(&lB3[buf][0][0], lane);
    #pragma unroll
    for (int s = 0; s < 8; ++s) {
      v16h nx1 = b1, nx3 = b3;
      if (s < 7) {                         // preload next subtile's fragments
        nx1 = load_b_frag(&lB1[buf][(s + 1) * 16][0], lane);
        nx3 = load_b_frag(&lB3[buf][(s + 1) * 16][0], lane);
      }
      accG[s] = WMMA_F16(af, b1, accG[s]);
      accU[s] = WMMA_F16(af, b3, accU[s]);
      b1 = nx1; b3 = nx3;
    }
  }

  // Epilogue: per-channel scale + SiLU*up in f32, store f16.
  // C layout: lane L, VGPR r -> M = r + 8*(L/16), N = L%16.
  #pragma unroll
  for (int s = 0; s < 8; ++s) {
    const int n  = n0 + s * 16 + (lane & 15);
    const float sg = s1[n];
    const float su = s3[n];
    const int mBase = m0 + wave * 16 + ((lane >> 4) * 8);
    #pragma unroll
    for (int r = 0; r < 8; ++r) {
      const float g   = accG[s][r] * sg;
      const float u   = accU[s][r] * su;
      const float sig = 1.0f / (1.0f + __expf(-g));
      h[(size_t)(mBase + r) * FFN + n] = (_Float16)(g * sig * u);
    }
  }
}

// ---------------------------------------------------------------------------
// Kernel 2: out[m,n] = s2[n] * sum_k h[m,k]*(q2[n,k]-128), fp32 out.
// Same double-buffered single-barrier structure, one weight matrix.
// ---------------------------------------------------------------------------
__global__ void __launch_bounds__(256)
down_kernel(const _Float16* __restrict__ h,
            const int* __restrict__ w2q, const float* __restrict__ s2,
            float* __restrict__ out)
{
  __shared__ _Float16 lA[2][BM][LDK];
  __shared__ _Float16 lB[2][BN][LDK];

  const int tid  = threadIdx.x;
  const int lane = tid & 31;
  const int wave = tid >> 5;
  const int m0   = blockIdx.x * BM;
  const int n0   = blockIdx.y * BN;

  const int arow = tid >> 2;              // 0..63 (2 passes over 128 rows)
  const int acol = (tid & 3) * 8;         // 16B chunks of f16
  const int brow = tid >> 3;              // 0..31 (4 passes)
  const int bcol = (tid & 7) * 4;

  const _Float16* ph = h   + (size_t)(m0 + arow) * FFN + acol;
  const int*      pb = w2q + (size_t)(n0 + brow) * FFN + bcol;
  const size_t astep = (size_t)64 * FFN;
  const size_t bstep = (size_t)32 * FFN;

  v8f acc[8] = {};
  int4 rb[4];

  auto issue_async = [&](int buf, int k0) {
    #pragma unroll
    for (int p = 0; p < 2; ++p)
      async_copy_b128(ph + p * astep + k0, &lA[buf][arow + p * 64][acol]);
  };

  issue_async(0, 0);
  #pragma unroll
  for (int p = 0; p < 4; ++p) rb[p] = *(const int4*)(pb + p * bstep);

  const int NK = FFN / BK;
  for (int kt = 0; kt < NK; ++kt) {
    const int buf = kt & 1;
    #pragma unroll
    for (int p = 0; p < 4; ++p)
      cvt_store_q(&lB[buf][brow + p * 32][bcol], rb[p]);

    if (kt + 1 < NK) {
      const int k0 = (kt + 1) * BK;
      issue_async(buf ^ 1, k0);           // streams while WMMAs run on buf
      #pragma unroll
      for (int p = 0; p < 4; ++p) {
        rb[p] = *(const int4*)(pb + p * bstep + k0);
        __builtin_prefetch(pb + p * bstep + k0 + BK, 0, 1);
      }
      asm volatile("s_wait_asynccnt 0x2" ::: "memory");
    } else {
      asm volatile("s_wait_asynccnt 0x0" ::: "memory");
    }
    __syncthreads();

    const v16h af = load_a_frag(&lA[buf][wave * 16][0], lane);
    v16h bf = load_b_frag(&lB[buf][0][0], lane);
    #pragma unroll
    for (int s = 0; s < 8; ++s) {
      v16h nb = bf;
      if (s < 7) nb = load_b_frag(&lB[buf][(s + 1) * 16][0], lane);
      acc[s] = WMMA_F16(af, bf, acc[s]);
      bf = nb;
    }
  }

  #pragma unroll
  for (int s = 0; s < 8; ++s) {
    const int n = n0 + s * 16 + (lane & 15);
    const float sc = s2[n];
    const int mBase = m0 + wave * 16 + ((lane >> 4) * 8);
    #pragma unroll
    for (int r = 0; r < 8; ++r)
      out[(size_t)(mBase + r) * HIDDEN + n] = acc[s][r] * sc;
  }
}

// ---------------------------------------------------------------------------
extern "C" void kernel_launch(void* const* d_in, const int* in_sizes, int n_in,
                              void* d_out, int out_size, void* d_ws, size_t ws_size,
                              hipStream_t stream) {
  const float* x   = (const float*)d_in[0];
  const int*   w1q = (const int*)d_in[1];
  const float* s1  = (const float*)d_in[2];
  const int*   w3q = (const int*)d_in[3];
  const float* s3  = (const float*)d_in[4];
  const int*   w2q = (const int*)d_in[5];
  const float* s2  = (const float*)d_in[6];
  float*     out  = (float*)d_out;

  // Workspace layout: [h f16: 2048*14336] [xh f16: 2048*4096]  (75.5 MB)
  _Float16* hbuf = (_Float16*)d_ws;
  _Float16* xh   = hbuf + (size_t)Q_LEN * FFN;

  // One-time fp32 -> f16 conversion of activations (K-invariant hoist).
  cvt_x_kernel<<<(Q_LEN * HIDDEN) / (256 * 8), 256, 0, stream>>>(x, xh);

  // M is the fast grid dimension so concurrently-resident blocks share the
  // streamed weight slice through the 192 MB L2 (weights read ~once from HBM).
  dim3 g1(Q_LEN / BM, FFN / BN);           // (16, 112)
  gateup_kernel<<<g1, 256, 0, stream>>>(xh, w1q, s1, w3q, s3, hbuf);

  dim3 g2(Q_LEN / BM, HIDDEN / BN);        // (16, 32)
  down_kernel<<<g2, 256, 0, stream>>>(hbuf, w2q, s2, out);
}
